// RCNN_68693706932253
// MI455X (gfx1250) — matrix operand order repaired and checked by reading, not compiled, not run on hardware
//
#include <hip/hip_runtime.h>
#include <hip/hip_bf16.h>

typedef __bf16 bf16_t;
typedef __attribute__((ext_vector_type(16))) __bf16 v16bf;
typedef __attribute__((ext_vector_type(8)))  __bf16 v8bf;
typedef __attribute__((ext_vector_type(8)))  float  v8f;
typedef __attribute__((ext_vector_type(4)))  float  v4f;

#define S_LEN 2048
#define BATCH 64
#define EDIM  128
#define HDIM  128
#define MDIM  256
#define CATD  384
#define CDIM  5

// ---------------- WMMA fragment helpers (CDNA5 wave32 layouts) ----------------

// A fragment: 16x32 bf16 tile from row-major src (leading dim = ld elements).
// lane(0-15): row m=lane, K=0..15; lane(16-31): K=16..31
// => two contiguous 16-byte loads per lane.
__device__ __forceinline__ v16bf load_a_frag(const bf16_t* src, int ld) {
  int lane = threadIdx.x & 31;
  int m = lane & 15;
  int half = lane >> 4;
  const bf16_t* base = src + (size_t)m * ld + 16 * half;
  v8bf lo = *(const v8bf*)(base);
  v8bf hi = *(const v8bf*)(base + 8);
  v16bf a;
#pragma unroll
  for (int i = 0; i < 8; ++i) { a[i] = lo[i]; a[i + 8] = hi[i]; }
  return a;
}

// B fragment: B = W^T chunk (32x16), W row-major (n rows, k cols), caller pre-offsets to (n0,k0).
// lanes 0-15: col n=lane, K=0..15; lanes 16-31: col n=lane-16, K=16..31 => one 32-byte load/lane.
__device__ __forceinline__ v16bf load_b_frag(const bf16_t* Wbase, int ldk) {
  int lane = threadIdx.x & 31;
  int n = lane & 15;
  int half = lane >> 4;
  return *(const v16bf*)(Wbase + (size_t)n * ldk + 16 * half);
}

__device__ __forceinline__ v8f wmma_bf16(v16bf a, v16bf b, v8f c) {
  return __builtin_amdgcn_wmma_f32_16x16x32_bf16(false, a, false, b, (short)0, c, false, false);
}

// Fragment-major storage layout for the scan inputs: per (s, n-tile jn, m-tile mt, lane)
// the 8 accumulator-row values are contiguous (16 bytes -> single b128 access).
__device__ __forceinline__ size_t efrag_off(int s, int jn, int mt, int lane) {
  return ((((size_t)s * 8 + jn) * 4 + mt) * 32 + lane) * 8;
}

__device__ __forceinline__ void atomicMaxF(float* addr, float v) {
  unsigned int* ua = (unsigned int*)addr;
  unsigned int old = *ua;
  while (true) {
    float f = __uint_as_float(old);
    if (f >= v) break;
    unsigned int assumed = old;
    old = atomicCAS(ua, assumed, __float_as_uint(v));
    if (old == assumed) break;
  }
}

// ---------------- Kernel 0: weight fp32->bf16 conversion + ymax init ----------------
__global__ void prep_kernel(const float* Wsl, const float* Wsr, const float* Wl, const float* Wr,
                            const float* Wmax,
                            bf16_t* wsl, bf16_t* wsr, bf16_t* wl, bf16_t* wr, bf16_t* wmax,
                            float* ymax) {
  int id = blockIdx.x * blockDim.x + threadIdx.x;
  if (id < HDIM * EDIM) {
    wsl[id] = (bf16_t)Wsl[id];
    wsr[id] = (bf16_t)Wsr[id];
    wl[id]  = (bf16_t)Wl[id];
    wr[id]  = (bf16_t)Wr[id];
  }
  if (id < MDIM * CATD) wmax[id] = (bf16_t)Wmax[id];
  if (id < BATCH * MDIM) ymax[id] = -2.0f;
}

// ---------------- Kernel 1: embedding gather + both input projections ----------------
// one block per sequence position s: 64 rows, N=128, K=128, for both W_sl and W_sr.
// Projections are written in fragment-major layout for the scan kernel.
__global__ void __launch_bounds__(256)
gather_proj_kernel(const int* inp, const float* emb_table,
                   const bf16_t* wsl, const bf16_t* wsr,
                   const float* bsl, const float* bsr,
                   bf16_t* emb_bf, bf16_t* embls, bf16_t* embrs) {
  __shared__ bf16_t Atile[BATCH * EDIM];
  __shared__ int stok[BATCH];
  int tid = threadIdx.x;
  size_t n0 = (size_t)blockIdx.x * BATCH;
  if (tid < BATCH) stok[tid] = inp[n0 + tid];
  __syncthreads();
#pragma unroll
  for (int i = 0; i < (BATCH * EDIM) / (256 * 8); ++i) {   // 8 elems per thread/iter
    int vidx = tid + i * 256;
    int idx = vidx * 8;
    int r = idx >> 7, e = idx & 127;
    const float* srcp = emb_table + (size_t)stok[r] * EDIM + e;
    v4f f0 = *(const v4f*)(srcp);
    v4f f1 = *(const v4f*)(srcp + 4);
    v8bf pk;
#pragma unroll
    for (int j = 0; j < 4; ++j) { pk[j] = (bf16_t)f0[j]; pk[4 + j] = (bf16_t)f1[j]; }
    *(v8bf*)(Atile + idx) = pk;
    *(v8bf*)(emb_bf + n0 * EDIM + idx) = pk;
  }
  __syncthreads();

  int w = tid >> 5;
  int p = w >> 2;          // 0: left proj, 1: right proj
  int mt = w & 3;          // M tile (16 rows)
  const bf16_t* W = p ? wsr : wsl;
  const float* bias = p ? bsr : bsl;
  bf16_t* dst = p ? embrs : embls;
  int lane = tid & 31, col = lane & 15;

  v16bf afr[4];
#pragma unroll
  for (int kc = 0; kc < 4; ++kc)
    afr[kc] = load_a_frag(Atile + mt * 16 * EDIM + kc * 32, EDIM);

#pragma unroll
  for (int jn = 0; jn < 8; ++jn) {
    float bv = bias[jn * 16 + col];
    v8f acc;
#pragma unroll
    for (int r = 0; r < 8; ++r) acc[r] = bv;
#pragma unroll
    for (int kc = 0; kc < 4; ++kc) {
      v16bf bfr = load_b_frag(W + (size_t)(jn * 16) * EDIM + kc * 32, EDIM);
      acc = wmma_bf16(afr[kc], bfr, acc);
    }
    v8bf pk;
#pragma unroll
    for (int r = 0; r < 8; ++r) pk[r] = (bf16_t)acc[r];
    *(v8bf*)(dst + efrag_off(blockIdx.x, jn, mt, lane)) = pk;  // one b128 store
  }
}

// ---------------- Kernel 2: the two sequential recurrences ----------------
// block 0: forward scan (W_l), block 1: backward scan (W_r).
// Weight matrix resident in VGPRs as 16 WMMA B-fragments; e-inputs software-pipelined.
__global__ void __launch_bounds__(256)
scan_kernel(const bf16_t* wl, const bf16_t* wr,
            const float* bl, const float* br,
            const float* cl0, const float* cr0,
            const bf16_t* embls, const bf16_t* embrs,
            bf16_t* cls, bf16_t* crs) {
  __shared__ bf16_t cstate[2][BATCH * HDIM];
  int rev = blockIdx.x;
  const bf16_t* W    = rev ? wr : wl;
  const float* bias  = rev ? br : bl;
  const float* c0    = rev ? cr0 : cl0;
  const bf16_t* esrc = rev ? embrs : embls;
  bf16_t* cout       = rev ? crs : cls;

  int tid = threadIdx.x;
#pragma unroll
  for (int i = 0; i < (BATCH * HDIM) / 256; ++i) {
    int idx = tid + i * 256;
    cstate[0][idx] = (bf16_t)c0[idx & (HDIM - 1)];
  }
  __syncthreads();

  int w = tid >> 5;
  int mt = w & 3;            // M tile
  int j0 = (w >> 2) * 4;     // first of 4 N tiles for this wave
  int lane = tid & 31, col = lane & 15, half = lane >> 4;

  v16bf Bf[16];              // W resident in VGPRs across all 2048 steps
#pragma unroll
  for (int jn = 0; jn < 4; ++jn)
#pragma unroll
    for (int kc = 0; kc < 4; ++kc)
      Bf[jn * 4 + kc] = load_b_frag(W + (size_t)((j0 + jn) * 16) * HDIM + kc * 32, HDIM);

  float bcol[4];
#pragma unroll
  for (int jn = 0; jn < 4; ++jn) bcol[jn] = bias[(j0 + jn) * 16 + col];

  // software-pipelined e fragments (one b128 per N-tile per step)
  v8bf ecur[4];
  {
    int se0 = rev ? (S_LEN - 1) : 0;
#pragma unroll
    for (int jn = 0; jn < 4; ++jn)
      ecur[jn] = *(const v8bf*)(esrc + efrag_off(se0, j0 + jn, mt, lane));
  }

  int pb = 0;
  for (int s = 0; s < S_LEN; ++s) {
    int se = rev ? (S_LEN - 1 - s) : s;
    // issue next step's e loads early (independent of LDS state)
    v8bf enext[4];
    if (s + 1 < S_LEN) {
      int sn = rev ? (S_LEN - 2 - s) : (s + 1);
#pragma unroll
      for (int jn = 0; jn < 4; ++jn)
        enext[jn] = *(const v8bf*)(esrc + efrag_off(sn, j0 + jn, mt, lane));
    }

    v16bf afr[4];
#pragma unroll
    for (int kc = 0; kc < 4; ++kc)
      afr[kc] = load_a_frag(&cstate[pb][mt * 16 * HDIM + kc * 32], HDIM);

#pragma unroll
    for (int jn = 0; jn < 4; ++jn) {
      int ncol0 = (j0 + jn) * 16;
      v8f acc;
#pragma unroll
      for (int r = 0; r < 8; ++r) acc[r] = bcol[jn] + (float)ecur[jn][r];
#pragma unroll
      for (int kc = 0; kc < 4; ++kc)
        acc = wmma_bf16(afr[kc], Bf[jn * 4 + kc], acc);
#pragma unroll
      for (int r = 0; r < 8; ++r) {
        int row = mt * 16 + r + 8 * half;
        bf16_t t = (bf16_t)tanhf(acc[r]);
        cstate[pb ^ 1][row * HDIM + ncol0 + col] = t;
        cout[((size_t)se * BATCH + row) * HDIM + ncol0 + col] = t;
      }
    }
    __syncthreads();
    pb ^= 1;
#pragma unroll
    for (int jn = 0; jn < 4; ++jn) ecur[jn] = enext[jn];
  }
}

// ---------------- Kernel 3: cat GEMM (K=384, N=256) + tanh + max-over-S ----------------
__global__ void __launch_bounds__(256)
maxpool_kernel(const bf16_t* cls, const bf16_t* emb_bf, const bf16_t* crs,
               const bf16_t* wmax, const float* bmax, float* ymax) {
  __shared__ bf16_t cat[BATCH * CATD];
  int tid = threadIdx.x;
  size_t s = blockIdx.x;
#pragma unroll
  for (int i = 0; i < (BATCH * CATD) / (256 * 8); ++i) {   // v8bf copies; 128 % 8 == 0
    int vidx = tid + i * 256;
    int idx = vidx * 8;
    int r = idx / CATD, c = idx % CATD;
    size_t rowbase = (s * BATCH + r) * (size_t)HDIM;
    const bf16_t* srcp;
    if (c < HDIM)          srcp = cls + rowbase + c;
    else if (c < 2 * HDIM) srcp = emb_bf + rowbase + (c - HDIM);
    else                   srcp = crs + rowbase + (c - 2 * HDIM);
    *(v8bf*)(cat + idx) = *(const v8bf*)srcp;
  }
  __syncthreads();

  int w = tid >> 5;
  int mt = w & 3;
  int j0 = (w >> 2) * 8;     // 8 N tiles per wave (N=256 -> 16 tiles)
  int lane = tid & 31, col = lane & 15, half = lane >> 4;

  v16bf afr[12];
#pragma unroll
  for (int kc = 0; kc < 12; ++kc)
    afr[kc] = load_a_frag(cat + mt * 16 * CATD + kc * 32, CATD);

#pragma unroll
  for (int jn = 0; jn < 8; ++jn) {
    int ncol0 = (j0 + jn) * 16;
    float bv = bmax[ncol0 + col];
    v8f acc;
#pragma unroll
    for (int r = 0; r < 8; ++r) acc[r] = bv;
#pragma unroll
    for (int kc = 0; kc < 12; ++kc) {
      v16bf bfr = load_b_frag(wmax + (size_t)ncol0 * CATD + kc * 32, CATD);
      acc = wmma_bf16(afr[kc], bfr, acc);
    }
#pragma unroll
    for (int r = 0; r < 8; ++r) {
      int brow = mt * 16 + r + 8 * half;   // batch index (block == one s)
      atomicMaxF(&ymax[brow * MDIM + ncol0 + col], tanhf(acc[r]));
    }
  }
}

// ---------------- Kernel 4: tiny classifier head + softmax ----------------
__global__ void head_kernel(const float* ymax, const float* Wdoc, const float* bdoc, float* out) {
  int b = threadIdx.x;
  if (b >= BATCH) return;
  float logits[CDIM];
#pragma unroll
  for (int c = 0; c < CDIM; ++c) logits[c] = bdoc[c];
  for (int k = 0; k < MDIM; ++k) {
    float y = ymax[b * MDIM + k];
#pragma unroll
    for (int c = 0; c < CDIM; ++c) logits[c] += y * Wdoc[c * MDIM + k];
  }
  float mx = logits[0];
#pragma unroll
  for (int c = 1; c < CDIM; ++c) mx = fmaxf(mx, logits[c]);
  float sum = 0.f;
#pragma unroll
  for (int c = 0; c < CDIM; ++c) { logits[c] = __expf(logits[c] - mx); sum += logits[c]; }
#pragma unroll
  for (int c = 0; c < CDIM; ++c) out[b * CDIM + c] = logits[c] / sum;
}

// ---------------- Host launch ----------------
extern "C" void kernel_launch(void* const* d_in, const int* in_sizes, int n_in,
                              void* d_out, int out_size, void* d_ws, size_t ws_size,
                              hipStream_t stream) {
  const int*   inp  = (const int*)d_in[0];
  const float* embt = (const float*)d_in[1];
  const float* Wsl  = (const float*)d_in[2];
  const float* bsl  = (const float*)d_in[3];
  const float* Wsr  = (const float*)d_in[4];
  const float* bsr  = (const float*)d_in[5];
  const float* Wl   = (const float*)d_in[6];
  const float* bl   = (const float*)d_in[7];
  const float* Wr   = (const float*)d_in[8];
  const float* br   = (const float*)d_in[9];
  const float* cl0  = (const float*)d_in[10];
  const float* cr0  = (const float*)d_in[11];
  const float* Wmax = (const float*)d_in[12];
  const float* bmax = (const float*)d_in[13];
  const float* Wdoc = (const float*)d_in[14];
  const float* bdoc = (const float*)d_in[15];

  char* ws = (char*)d_ws;
  size_t off = 0;
  auto take = [&](size_t bytes) -> char* {
    char* p = ws + off;
    off = (off + bytes + 255) & ~(size_t)255;
    return p;
  };
  bf16_t* wsl_bf  = (bf16_t*)take((size_t)HDIM * EDIM * 2);
  bf16_t* wsr_bf  = (bf16_t*)take((size_t)HDIM * EDIM * 2);
  bf16_t* wl_bf   = (bf16_t*)take((size_t)HDIM * HDIM * 2);
  bf16_t* wr_bf   = (bf16_t*)take((size_t)HDIM * HDIM * 2);
  bf16_t* wmax_bf = (bf16_t*)take((size_t)MDIM * CATD * 2);
  bf16_t* emb_bf  = (bf16_t*)take((size_t)S_LEN * BATCH * EDIM * 2);
  bf16_t* embls   = (bf16_t*)take((size_t)S_LEN * BATCH * HDIM * 2);   // fragment-major
  bf16_t* embrs   = (bf16_t*)take((size_t)S_LEN * BATCH * HDIM * 2);   // fragment-major
  bf16_t* cls     = (bf16_t*)take((size_t)S_LEN * BATCH * HDIM * 2);
  bf16_t* crs     = (bf16_t*)take((size_t)S_LEN * BATCH * HDIM * 2);
  float*  ymax    = (float*)take((size_t)BATCH * MDIM * 4);

  prep_kernel<<<(MDIM * CATD + 255) / 256, 256, 0, stream>>>(
      Wsl, Wsr, Wl, Wr, Wmax, wsl_bf, wsr_bf, wl_bf, wr_bf, wmax_bf, ymax);

  gather_proj_kernel<<<S_LEN, 256, 0, stream>>>(
      inp, embt, wsl_bf, wsr_bf, bsl, bsr, emb_bf, embls, embrs);

  scan_kernel<<<2, 256, 0, stream>>>(
      wl_bf, wr_bf, bl, br, cl0, cr0, embls, embrs, cls, crs);

  maxpool_kernel<<<S_LEN, 256, 0, stream>>>(
      cls, emb_bf, crs, wmax_bf, bmax, ymax);

  head_kernel<<<1, 64, 0, stream>>>(ymax, Wdoc, bdoc, (float*)d_out);
}
